// CrossAttention_58239756534313
// MI455X (gfx1250) — matrix operand order repaired and checked by reading, not compiled, not run on hardware
//
#include <hip/hip_runtime.h>
#include <math.h>

// ---------------------------------------------------------------------------
// CrossAttention on gfx1250 (MI455X): bf16 WMMA GEMMs + flash attention.
// Pipeline: q=x@Wq, k=ctx@Wk, v=ctx@Wv  (bf16 out) -> flash attn (bf16 out)
//           -> out = O@Wo + bo (f32 out).
// CDNA5 paths: v_wmma_f32_16x16x32_bf16, global_load_async_to_lds_b64 +
// s_wait_asynccnt (guarded by __has_builtin), global_prefetch_b8.
// ---------------------------------------------------------------------------

typedef __attribute__((ext_vector_type(16))) __bf16 v16bf;
typedef __attribute__((ext_vector_type(8)))  float  v8f;
typedef __attribute__((ext_vector_type(2)))  int    v2i;

#define HEADS  8
#define DIMH   64
#define INNERD 512
#define DMODEL 1024
#define NQ     4096
#define MKEYS  4096
#define BATCH  2

#if defined(__has_builtin)
#if __has_builtin(__builtin_amdgcn_global_load_async_to_lds_b64) && \
    __has_builtin(__builtin_amdgcn_s_wait_asynccnt)
#define USE_ASYNC_LDS 1
#endif
#endif

#ifdef USE_ASYNC_LDS
// b64 variant moves one v2i (64 bits) per lane: global as(1) -> LDS as(3)
#define AS1V2(p) ((__attribute__((address_space(1))) v2i*)(p))
#define AS3V2(p) ((__attribute__((address_space(3))) v2i*)(p))
#endif

// A-fragment (16x32 bf16, K inner): lane holds row m=lane&15,
// elements = row[kb .. kb+7] ++ row[16+kb .. 16+kb+7], kb = 8*(lane>=16).
__device__ __forceinline__ v16bf ld_a_frag(const __bf16* __restrict__ row, int hi) {
  v16bf f;
  const int kb = hi * 8;
#pragma unroll
  for (int i = 0; i < 8; ++i) {
    f[i]     = row[kb + i];
    f[8 + i] = row[16 + kb + i];
  }
  return f;
}

// B-fragment (32x16 bf16): lane holds column n=lane&15,
// elements = col[kb .. kb+7] ++ col[16+kb .. 16+kb+7], kb = 8*(lane>=16). `col` points at B^T[n][0] (K contiguous).
__device__ __forceinline__ v16bf ld_b_frag(const __bf16* __restrict__ col, int hi) {
  v16bf f;
  const int kb = hi * 8;
#pragma unroll
  for (int i = 0; i < 8; ++i) {
    f[i]     = col[kb + i];
    f[8 + i] = col[16 + kb + i];
  }
  return f;
}

// ---------------------------------------------------------------------------
// Tiled GEMM: C[M,N] = A[M,K] * B[K,N] (+bias). Block tile 128x64, K-step 32.
// 256 threads = 8 waves; each wave computes a 32x32 tile (2x2 WMMA accums).
// AT=float: A staged via float4 + cvt.  AT=__bf16: A staged via async-to-LDS.
// ---------------------------------------------------------------------------
template <typename AT, bool OUT_F32, bool BIAS>
__global__ __launch_bounds__(256) void gemm_wmma(
    const AT* __restrict__ A, const float* __restrict__ B,
    void* __restrict__ Cout, const float* __restrict__ bias,
    int Msz, int Nsz, int Ksz) {
  __shared__ __bf16 As[128][40];   // [row][k], padded (row stride 80B, 8-aligned)
  __shared__ __bf16 Bt[64][40];    // [col][k] (B transposed), padded

  const int tid  = threadIdx.x;
  const int lane = tid & 31, w = tid >> 5;
  const int hi   = lane >> 4, nl = lane & 15;
  const int m0   = blockIdx.x * 128, n0 = blockIdx.y * 64;
  const int wm   = (w >> 1) * 32,   wn = (w & 1) * 32;

  v8f acc[2][2] = {};

  for (int kt = 0; kt < Ksz; kt += 32) {
    // ---- stage A tile (128 x 32) ----
    if constexpr (sizeof(AT) == 4) {
      // f32 -> bf16 convert through VGPRs; b128 global reads
#pragma unroll
      for (int j = 0; j < 4; ++j) {
        int lin = tid + j * 256;             // 1024 float4 chunks
        int r = lin >> 3, c4 = lin & 7;
        const float4 val = *(const float4*)&A[(size_t)(m0 + r) * Ksz + kt + c4 * 4];
        As[r][c4 * 4 + 0] = (__bf16)val.x;
        As[r][c4 * 4 + 1] = (__bf16)val.y;
        As[r][c4 * 4 + 2] = (__bf16)val.z;
        As[r][c4 * 4 + 3] = (__bf16)val.w;
      }
    } else {
      // bf16 -> bf16 straight copy
#ifdef USE_ASYNC_LDS
#pragma unroll
      for (int j = 0; j < 4; ++j) {
        int lin = tid + j * 256;             // 1024 b64 chunks (4 bf16 each)
        int r = lin >> 3, c4 = lin & 7;
        __builtin_amdgcn_global_load_async_to_lds_b64(
            AS1V2(&A[(size_t)(m0 + r) * Ksz + kt + c4 * 4]),
            AS3V2(&As[r][c4 * 4]), 0, 0);
      }
#else
#pragma unroll
      for (int j = 0; j < 2; ++j) {
        int lin = tid + j * 256;             // 512 b128 chunks (8 bf16 each)
        int r = lin >> 2, c8 = lin & 3;
        union { uint4 u4; __bf16 h[8]; } av;
        av.u4 = *(const uint4*)&A[(size_t)(m0 + r) * Ksz + kt + c8 * 8];
#pragma unroll
        for (int e = 0; e < 8; ++e) As[r][c8 * 8 + e] = av.h[e];
      }
#endif
    }
    // ---- stage B tile (32 x 64), transposed; b128 global reads ----
#pragma unroll
    for (int j = 0; j < 2; ++j) {
      int lin = tid + j * 256;               // 512 float4 chunks
      int kr = lin >> 4, c4 = lin & 15;
      const float4 val = *(const float4*)&B[(size_t)(kt + kr) * Nsz + n0 + c4 * 4];
      Bt[c4 * 4 + 0][kr] = (__bf16)val.x;
      Bt[c4 * 4 + 1][kr] = (__bf16)val.y;
      Bt[c4 * 4 + 2][kr] = (__bf16)val.z;
      Bt[c4 * 4 + 3][kr] = (__bf16)val.w;
    }
    if (kt + 32 < Ksz)
      __builtin_prefetch((const void*)&A[(size_t)(m0 + (tid >> 3)) * Ksz + kt + 32], 0, 0);
#ifdef USE_ASYNC_LDS
    if constexpr (sizeof(AT) == 2) __builtin_amdgcn_s_wait_asynccnt(0);
#endif
    __syncthreads();

    v16bf af[2], bfm[2];
#pragma unroll
    for (int rm = 0; rm < 2; ++rm) af[rm]  = ld_a_frag(&As[wm + rm * 16 + nl][0], hi);
#pragma unroll
    for (int cn = 0; cn < 2; ++cn) bfm[cn] = ld_b_frag(&Bt[wn + cn * 16 + nl][0], hi);

#pragma unroll
    for (int rm = 0; rm < 2; ++rm)
#pragma unroll
      for (int cn = 0; cn < 2; ++cn)
        acc[rm][cn] = __builtin_amdgcn_wmma_f32_16x16x32_bf16(
            false, af[rm], false, bfm[cn], (short)0, acc[rm][cn], false, false);
    __syncthreads();
  }

  // C layout: VGPR r -> row (hi*8 + r), column = lane&15 within 16-wide tile
#pragma unroll
  for (int rm = 0; rm < 2; ++rm)
#pragma unroll
    for (int cn = 0; cn < 2; ++cn)
#pragma unroll
      for (int r = 0; r < 8; ++r) {
        int row = m0 + wm + rm * 16 + hi * 8 + r;
        int col = n0 + wn + cn * 16 + nl;
        float val = acc[rm][cn][r];
        if constexpr (BIAS) val += bias[col];
        if constexpr (OUT_F32)
          ((float*)Cout)[(size_t)row * Nsz + col] = val;
        else
          ((__bf16*)Cout)[(size_t)row * Nsz + col] = (__bf16)val;
      }
}

// ---------------------------------------------------------------------------
// Flash attention: block = 128 query rows of one (batch, head); 8 waves,
// each wave owns 16 query rows. Streams 64-key tiles of K/V through LDS,
// keeps O accumulator + online-softmax stats in registers.
// K tile: async global->LDS DMA. V tile: b128 reads + LDS transpose scatter.
// ---------------------------------------------------------------------------
__global__ __launch_bounds__(256) void flash_attn(
    const __bf16* __restrict__ q, const __bf16* __restrict__ k,
    const __bf16* __restrict__ v, __bf16* __restrict__ o) {
  __shared__ __bf16 Ks[64][72];    // [key][dh]   (row stride 144B, 8-aligned)
  __shared__ __bf16 Vt[64][72];    // [dh][key]   (transposed for B-frags)
  __shared__ __bf16 Ps[128][72];   // P (C-layout -> A-layout relayout buffer)

  const int tid  = threadIdx.x;
  const int lane = tid & 31, w = tid >> 5;
  const int hi   = lane >> 4, nl = lane & 15;
  const int b = blockIdx.z, h = blockIdx.y;
  const int n0 = blockIdx.x * 128;
  const float scale = 0.125f;  // 64^-0.5

  // Q fragments (16 rows x 64 dh) pinned in registers for the whole K loop
  const __bf16* qrow = q + ((size_t)(b * NQ + n0 + w * 16 + nl)) * INNERD + h * DIMH;
  v16bf qa[2];
#pragma unroll
  for (int kc = 0; kc < 2; ++kc) qa[kc] = ld_a_frag(qrow + kc * 32, hi);

  v8f oacc[4] = {};
  float mrun[8], lrun[8];
#pragma unroll
  for (int r = 0; r < 8; ++r) { mrun[r] = -1e30f; lrun[r] = 0.f; }

  for (int kt = 0; kt < MKEYS; kt += 64) {
    const size_t kvbase = ((size_t)(b * MKEYS + kt)) * INNERD + h * DIMH;

    // ---- stage K tile (straight copy) ----
#ifdef USE_ASYNC_LDS
#pragma unroll
    for (int j = 0; j < 4; ++j) {
      int lin = tid + j * 256;               // 1024 b64 chunks (4 bf16 each)
      int key = lin >> 4, c4 = lin & 15;
      __builtin_amdgcn_global_load_async_to_lds_b64(
          AS1V2(&k[kvbase + (size_t)key * INNERD + c4 * 4]),
          AS3V2(&Ks[key][c4 * 4]), 0, 0);
    }
#else
#pragma unroll
    for (int j = 0; j < 2; ++j) {
      int lin = tid + j * 256;               // 512 b128 chunks (8 bf16 each)
      int key = lin >> 3, c8 = lin & 7;
      union { uint4 u4; __bf16 hh[8]; } kk;
      kk.u4 = *(const uint4*)&k[kvbase + (size_t)key * INNERD + c8 * 8];
#pragma unroll
      for (int e = 0; e < 8; ++e) Ks[key][c8 * 8 + e] = kk.hh[e];
    }
#endif
    // ---- stage V tile transposed (b128 reads, scalar LDS scatter) ----
#pragma unroll
    for (int j = 0; j < 2; ++j) {
      int lin = tid + j * 256;               // 512 b128 chunks
      int key = lin >> 3, c8 = lin & 7;
      union { uint4 u4; __bf16 hh[8]; } vv;
      vv.u4 = *(const uint4*)&v[kvbase + (size_t)key * INNERD + c8 * 8];
#pragma unroll
      for (int e = 0; e < 8; ++e) Vt[c8 * 8 + e][key] = vv.hh[e];
    }
    if (kt + 64 < MKEYS) {
      __builtin_prefetch((const void*)(k + kvbase + (size_t)64 * INNERD), 0, 0);
      __builtin_prefetch((const void*)(v + kvbase + (size_t)64 * INNERD), 0, 0);
    }
#ifdef USE_ASYNC_LDS
    __builtin_amdgcn_s_wait_asynccnt(0);
#endif
    __syncthreads();

    // S = Q K^T : 4 column tiles x (dh chunks of 32) -> 8 WMMA
    v8f s[4];
#pragma unroll
    for (int t = 0; t < 4; ++t) {
      v8f z = {};
#pragma unroll
      for (int kc = 0; kc < 2; ++kc) {
        v16bf bk = ld_b_frag(&Ks[t * 16 + nl][kc * 32], hi);
        z = __builtin_amdgcn_wmma_f32_16x16x32_bf16(false, qa[kc], false, bk,
                                                    (short)0, z, false, false);
      }
      s[t] = z;
    }

    // online softmax: row stats via xor-shuffle over the 16-lane half
#pragma unroll
    for (int r = 0; r < 8; ++r) {
      float rmax = -1e30f;
#pragma unroll
      for (int t = 0; t < 4; ++t) { s[t][r] *= scale; rmax = fmaxf(rmax, s[t][r]); }
#pragma unroll
      for (int msk = 8; msk >= 1; msk >>= 1) rmax = fmaxf(rmax, __shfl_xor(rmax, msk, 32));
      float mnew = fmaxf(mrun[r], rmax);
      float fac  = __expf(mrun[r] - mnew);
      float rsum = 0.f;
#pragma unroll
      for (int t = 0; t < 4; ++t) {
        float p = __expf(s[t][r] - mnew);
        s[t][r] = p;
        rsum += p;
      }
#pragma unroll
      for (int msk = 8; msk >= 1; msk >>= 1) rsum += __shfl_xor(rsum, msk, 32);
      lrun[r] = lrun[r] * fac + rsum;
      mrun[r] = mnew;
#pragma unroll
      for (int t = 0; t < 4; ++t) oacc[t][r] *= fac;
    }

    // P: C-layout -> LDS -> A-layout
#pragma unroll
    for (int t = 0; t < 4; ++t)
#pragma unroll
      for (int r = 0; r < 8; ++r)
        Ps[w * 16 + hi * 8 + r][t * 16 + nl] = (__bf16)s[t][r];
    __syncthreads();

    // O += P V : 4 dh tiles x (key chunks of 32) -> 8 WMMA
#pragma unroll
    for (int kc = 0; kc < 2; ++kc) {
      v16bf pa = ld_a_frag(&Ps[w * 16 + nl][kc * 32], hi);
#pragma unroll
      for (int t = 0; t < 4; ++t) {
        v16bf bv = ld_b_frag(&Vt[t * 16 + nl][kc * 32], hi);
        oacc[t] = __builtin_amdgcn_wmma_f32_16x16x32_bf16(false, pa, false, bv,
                                                          (short)0, oacc[t], false, false);
      }
    }
    __syncthreads();
  }

  // normalize and store O (bf16, [b, n, h*64+dh] layout)
#pragma unroll
  for (int r = 0; r < 8; ++r) {
    float inv = 1.f / lrun[r];
    int row = n0 + w * 16 + hi * 8 + r;
#pragma unroll
    for (int t = 0; t < 4; ++t) {
      int col = h * DIMH + t * 16 + nl;
      o[((size_t)(b * NQ + row)) * INNERD + col] = (__bf16)(oacc[t][r] * inv);
    }
  }
}

// ---------------------------------------------------------------------------
extern "C" void kernel_launch(void* const* d_in, const int* in_sizes, int n_in,
                              void* d_out, int out_size, void* d_ws, size_t ws_size,
                              hipStream_t stream) {
  const float* x   = (const float*)d_in[0];
  const float* ctx = (const float*)d_in[1];
  const float* Wq  = (const float*)d_in[2];
  const float* Wk  = (const float*)d_in[3];
  const float* Wv  = (const float*)d_in[4];
  const float* Wo  = (const float*)d_in[5];
  const float* bo  = (const float*)d_in[6];
  float* out = (float*)d_out;

  const size_t QKV = (size_t)BATCH * NQ * INNERD;  // 4,194,304 bf16 each
  __bf16* qws = (__bf16*)d_ws;
  __bf16* kws = qws + QKV;
  __bf16* vws = kws + QKV;
  __bf16* ows = vws + QKV;

  dim3 blk(256);
  const int Mrows = BATCH * NQ;  // 8192

  // projections: bf16 outputs
  gemm_wmma<float, false, false><<<dim3(Mrows / 128, INNERD / 64), blk, 0, stream>>>(
      x, Wq, (void*)qws, nullptr, Mrows, INNERD, DMODEL);
  gemm_wmma<float, false, false><<<dim3(Mrows / 128, INNERD / 64), blk, 0, stream>>>(
      ctx, Wk, (void*)kws, nullptr, Mrows, INNERD, DMODEL);
  gemm_wmma<float, false, false><<<dim3(Mrows / 128, INNERD / 64), blk, 0, stream>>>(
      ctx, Wv, (void*)vws, nullptr, Mrows, INNERD, DMODEL);

  // flash attention
  flash_attn<<<dim3(NQ / 128, HEADS, BATCH), blk, 0, stream>>>(qws, kws, vws, ows);

  // output projection + bias: f32 output
  gemm_wmma<__bf16, true, true><<<dim3(Mrows / 128, DMODEL / 64), blk, 0, stream>>>(
      ows, Wo, (void*)out, bo, Mrows, DMODEL, INNERD);
}
